// CrossModalGAE_30932354465913
// MI455X (gfx1250) — compile-verified, matrix operand order checked by reading, not run on hardware
//
#include <hip/hip_runtime.h>
#include <hip/hip_bf16.h>

// ---------------------------------------------------------------------------
// Types for CDNA5 WMMA (wave32): D(16x16,f32) = A(16x32,f16) x B(32x16,f16) + C
// ---------------------------------------------------------------------------
typedef _Float16 half_t;
typedef __attribute__((ext_vector_type(16))) _Float16 v16h;
typedef __attribute__((ext_vector_type(8)))  _Float16 v8h;
typedef __attribute__((ext_vector_type(4)))  _Float16 v4h;
typedef __attribute__((ext_vector_type(8)))  float    v8f;
typedef __attribute__((ext_vector_type(4)))  float    v4f;

#define WMMA_F16(a, b, c) \
  __builtin_amdgcn_wmma_f32_16x16x32_f16(false, (a), false, (b), (short)0, (c), false, false)

// Problem dims (fixed by the reference)
#define NN   6000
#define HH   512
#define LL   128
#define DD1  3000
#define DD2  1000
#define E1   102000   /* 96000 + 6000 self loops */
#define EC   204000   /* 192000 + 12000 self loops */

// Load one 16x32 f16 fragment half-pair from LDS (contraction dim contiguous).
// p points at the start of the 32-wide row; hs selects the lane-half K slice.
// Layout (ISA 7.12.2): elems 0..7 = K[hs*8 .. hs*8+7], elems 8..15 = K[16+hs*8 ..].
__device__ __forceinline__ v16h frag_ld(const half_t* p, int hs) {
  v8h lo = *(const v8h*)(p + hs * 8);
  v8h hi = *(const v8h*)(p + 16 + hs * 8);
  return __builtin_shufflevector(lo, hi, 0, 1, 2, 3, 4, 5, 6, 7,
                                 8, 9, 10, 11, 12, 13, 14, 15);
}

__device__ __forceinline__ v4h cvt4(v4f x) {
  v4h r;
  r[0] = (half_t)x[0]; r[1] = (half_t)x[1];
  r[2] = (half_t)x[2]; r[3] = (half_t)x[3];
  return r;
}

// ---------------------------------------------------------------------------
// Generic elementwise / fill kernels
// ---------------------------------------------------------------------------
__global__ void fill_f32_kernel(float* __restrict__ p, long n, float v) {
  long i = (long)blockIdx.x * blockDim.x + threadIdx.x;
  if (i < n) p[i] = v;
}

__device__ __forceinline__ unsigned enc_f(float f) {
  unsigned u = __float_as_uint(f);
  return (u & 0x80000000u) ? ~u : (u | 0x80000000u);
}
__device__ __forceinline__ float dec_f(unsigned u) {
  return (u & 0x80000000u) ? __uint_as_float(u ^ 0x80000000u) : __uint_as_float(~u);
}

__global__ void fill_enc_kernel(unsigned* __restrict__ p, long n, float v) {
  long i = (long)blockIdx.x * blockDim.x + threadIdx.x;
  if (i < n) p[i] = enc_f(v);
}

__device__ __forceinline__ float elu_f(float v) {
  return v > 0.f ? v : (__expf(v) - 1.f);
}

// mode 0: out = accA + bias ; mode 1: out = elu(accA + bias)
// mode 2: out = elu(0.5*accA + 0.5*accB + bias)
__global__ void gat_epilogue_kernel(const float* __restrict__ accA,
                                    const float* __restrict__ accB,
                                    const float* __restrict__ bias,
                                    float* __restrict__ out, long total, int d, int mode) {
  long i = (long)blockIdx.x * blockDim.x + threadIdx.x;
  if (i >= total) return;
  int c = (int)(i % d);
  float v;
  if (mode == 2) v = 0.5f * accA[i] + 0.5f * accB[i] + bias[c];
  else           v = accA[i] + bias[c];
  if (mode >= 1) v = elu_f(v);
  out[i] = v;
}

// out = elu(0.5*x + 0.5*y)
__global__ void combine_elu_kernel(const float* __restrict__ x, const float* __restrict__ y,
                                   float* __restrict__ out, long total) {
  long i = (long)blockIdx.x * blockDim.x + threadIdx.x;
  if (i >= total) return;
  out[i] = elu_f(0.5f * x[i] + 0.5f * y[i]);
}

// ---------------------------------------------------------------------------
// Row-wise l2 normalize: out = in / max(||in||, 1e-12)
// ---------------------------------------------------------------------------
__global__ __launch_bounds__(128) void l2norm_kernel(const float* __restrict__ in,
                                                     float* __restrict__ out, int d) {
  __shared__ float ss[128];
  int row = blockIdx.x;
  const float* rp = in + (long)row * d;
  float a = 0.f;
  for (int j = threadIdx.x; j < d; j += 128) { float v = rp[j]; a += v * v; }
  ss[threadIdx.x] = a;
  __syncthreads();
  for (int s = 64; s > 0; s >>= 1) {
    if (threadIdx.x < s) ss[threadIdx.x] += ss[threadIdx.x + s];
    __syncthreads();
  }
  float denom = fmaxf(sqrtf(ss[0]), 1e-12f);
  float* op = out + (long)row * d;
  for (int j = threadIdx.x; j < d; j += 128) op[j] = rp[j] / denom;
}

// ---------------------------------------------------------------------------
// GEMV pair: el = h @ al, er = h @ ar   (one block per row)
// ---------------------------------------------------------------------------
__global__ __launch_bounds__(128) void gemv_elr_kernel(const float* __restrict__ h,
                                                       const float* __restrict__ al,
                                                       const float* __restrict__ ar,
                                                       float* __restrict__ el,
                                                       float* __restrict__ er, int d) {
  __shared__ float sl[128], sr[128];
  int row = blockIdx.x;
  const float* rp = h + (long)row * d;
  float a = 0.f, b = 0.f;
  for (int j = threadIdx.x; j < d; j += 128) {
    float v = rp[j];
    a += v * al[j];
    b += v * ar[j];
  }
  sl[threadIdx.x] = a; sr[threadIdx.x] = b;
  __syncthreads();
  for (int s = 64; s > 0; s >>= 1) {
    if (threadIdx.x < s) { sl[threadIdx.x] += sl[threadIdx.x + s]; sr[threadIdx.x] += sr[threadIdx.x + s]; }
    __syncthreads();
  }
  if (threadIdx.x == 0) { el[row] = sl[0]; er[row] = sr[0]; }
}

// ---------------------------------------------------------------------------
// GAT edge-softmax passes (segment max / exp-sum / weighted scatter)
// ---------------------------------------------------------------------------
__device__ __forceinline__ float leaky_f(float e) { return e > 0.f ? e : 0.2f * e; }

__global__ void edge_max_kernel(const int* __restrict__ src, const int* __restrict__ dst,
                                const float* __restrict__ el, const float* __restrict__ er,
                                unsigned* __restrict__ m, int E) {
  int i = blockIdx.x * blockDim.x + threadIdx.x;
  if (i >= E) return;
  float e = leaky_f(el[src[i]] + er[dst[i]]);
  atomicMax(&m[dst[i]], enc_f(e));
}

__global__ void edge_expsum_kernel(const int* __restrict__ src, const int* __restrict__ dst,
                                   const float* __restrict__ el, const float* __restrict__ er,
                                   const unsigned* __restrict__ m, float* __restrict__ s,
                                   float* __restrict__ ex, int E) {
  int i = blockIdx.x * blockDim.x + threadIdx.x;
  if (i >= E) return;
  int d = dst[i];
  float e = leaky_f(el[src[i]] + er[d]);
  float v = __expf(e - dec_f(m[d]));
  ex[i] = v;
  atomicAdd(&s[d], v);
}

__global__ __launch_bounds__(256) void edge_scatter_kernel(const int* __restrict__ src,
                                                           const int* __restrict__ dst,
                                                           const float* __restrict__ ex,
                                                           const float* __restrict__ s,
                                                           const float* __restrict__ h,
                                                           float* __restrict__ out, int d) {
  int e = blockIdx.x;
  int si = src[e], di = dst[e];
  float a = ex[e] / s[di];
  const float* hp = h + (long)si * d;
  float* op = out + (long)di * d;
  for (int j = threadIdx.x; j < d; j += 256) atomicAdd(&op[j], hp[j] * a);
}

// ---------------------------------------------------------------------------
// WMMA GEMM: C[M,Nc] = A[M,K] @ B[K,Nc] (+ bias), f32 in/out, f16 compute.
// Tile 64x64, 8 waves; each wave: 1 M-tile x 2 N-tiles.
// As is row-major (M x K); Bs is stored K-major per column (transposed) so
// every fragment is two contiguous 16B LDS loads (ds_load_b128).
// ---------------------------------------------------------------------------
__global__ __launch_bounds__(256) void wmma_gemm_kernel(const float* __restrict__ A,
                                                        const float* __restrict__ B,
                                                        const float* __restrict__ bias,
                                                        float* __restrict__ C,
                                                        int M, int K, int Nc) {
  __shared__ __attribute__((aligned(16))) half_t As[64][40];  // [m][k], 32 + 8 pad
  __shared__ __attribute__((aligned(16))) half_t Bs[64][40];  // [n][k], 32 + 8 pad
  const int tid  = threadIdx.x;
  const int lane = tid & 31, wave = tid >> 5;
  const int hs = lane >> 4, ln = lane & 15;
  const int m0 = blockIdx.y * 64, n0 = blockIdx.x * 64;
  const int mt = wave & 3;            // M tile 0..3
  const int nh = wave >> 2;           // 0..1 -> N tiles 2*nh, 2*nh+1
  const int col0 = nh * 32 + ln;
  const int col1 = col0 + 16;
  const bool intA = (m0 + 64 <= M);
  const bool intB = (n0 + 64 <= Nc);

  v8f acc0 = {}; v8f acc1 = {};

  for (int k0 = 0; k0 < K; k0 += 32) {
    const bool kfull = (k0 + 32 <= K);
    __syncthreads();
    // ---- stage A tile (f32 -> f16), row-major ----
    if (intA && kfull) {
      #pragma unroll
      for (int i = 0; i < 2; i++) {
        int idx = tid + i * 256;                 // 0..511 vectors of 4
        int r = idx >> 3, c4 = (idx & 7) * 4;
        v4f f = *(const v4f*)&A[(long)(m0 + r) * K + k0 + c4];
        *(v4h*)&As[r][c4] = cvt4(f);
      }
    } else {
      #pragma unroll
      for (int i = 0; i < 8; i++) {
        int idx = tid + i * 256;
        int r = idx >> 5, c = idx & 31;
        int gm = m0 + r, gk = k0 + c;
        float v = (gm < M && gk < K) ? A[(long)gm * K + gk] : 0.f;
        As[r][c] = (half_t)v;
      }
    }
    // ---- stage B tile (f32 -> f16), transposed to [n][k] ----
    if (intB && kfull) {
      #pragma unroll
      for (int i = 0; i < 2; i++) {
        int idx = tid + i * 256;                 // 0..511 quads along k
        int c = idx & 63, r4 = (idx >> 6) * 4;
        const float* bp = &B[(long)(k0 + r4) * Nc + n0 + c];
        v4h h4;
        h4[0] = (half_t)bp[0];
        h4[1] = (half_t)bp[Nc];
        h4[2] = (half_t)bp[2 * (long)Nc];
        h4[3] = (half_t)bp[3 * (long)Nc];
        *(v4h*)&Bs[c][r4] = h4;
      }
    } else {
      #pragma unroll
      for (int i = 0; i < 8; i++) {
        int idx = tid + i * 256;
        int r = idx >> 6, c = idx & 63;
        int gk = k0 + r, gn = n0 + c;
        float v = (gk < K && gn < Nc) ? B[(long)gk * Nc + gn] : 0.f;
        Bs[c][r] = (half_t)v;
      }
    }
    if (kfull && k0 + 32 < K) {
      int gm = m0 + (tid >> 2);
      if (gm < M) __builtin_prefetch(&A[(long)gm * K + k0 + 32], 0, 1);
    }
    __syncthreads();

    v16h a  = frag_ld(&As[mt * 16 + ln][0], hs);
    v16h b0 = frag_ld(&Bs[col0][0], hs);
    v16h b1 = frag_ld(&Bs[col1][0], hs);
    acc0 = WMMA_F16(a, b0, acc0);
    acc1 = WMMA_F16(a, b1, acc1);
  }

  // store (C layout: lane -> n = ln, m = r + 8*hs)
  #pragma unroll
  for (int r = 0; r < 8; r++) {
    int gm  = m0 + mt * 16 + r + 8 * hs;
    int gn0 = n0 + nh * 32 + ln;
    int gn1 = gn0 + 16;
    if (gm < M) {
      if (gn0 < Nc) C[(long)gm * Nc + gn0] = acc0[r] + (bias ? bias[gn0] : 0.f);
      if (gn1 < Nc) C[(long)gm * Nc + gn1] = acc1[r] + (bias ? bias[gn1] : 0.f);
    }
  }
}

// ---------------------------------------------------------------------------
// Flash attention (masked), H fixed 512.
// O[i,:] = softmax_j(mask(i,j) ? (Q_i.K_j)/sqrt(H) : -inf) @ V
// transposed!=0 -> mask(i,j) = mask[j*n + i]  (for the scores.T direction)
// Block: 16 queries, 8 waves; wave w accumulates output dims [64w, 64w+64).
// ---------------------------------------------------------------------------
__global__ __launch_bounds__(256) void flash_attn_kernel(const float* __restrict__ Q,
                                                         const float* __restrict__ Km,
                                                         const float* __restrict__ V,
                                                         const unsigned char* __restrict__ mask,
                                                         float* __restrict__ O,
                                                         int n, int transposed) {
  __shared__ __attribute__((aligned(16))) half_t Qs[16][520];    // 16 x 512 (+8)
  __shared__ __attribute__((aligned(16))) half_t Ks[32][136];    // 32 x 128 (+8)
  __shared__ __attribute__((aligned(16))) half_t Ps[8][16][40];  // per-wave 16 x 32 (+8)

  const int tid  = threadIdx.x;
  const int lane = tid & 31, wave = tid >> 5;
  const int hs = lane >> 4, ln = lane & 15;
  const int q0 = blockIdx.x * 16;   // n % 16 == 0 -> queries always in range
  const float scale = 0.044194173824159216f;  // 1/sqrt(512)

  // stage Q tile (always interior)
  #pragma unroll
  for (int i = 0; i < 8; i++) {
    int idx = tid + i * 256;                    // 0..2047 vectors of 4
    int r = idx >> 7, c4 = (idx & 127) * 4;
    v4f f = *(const v4f*)&Q[(long)(q0 + r) * HH + c4];
    *(v4h*)&Qs[r][c4] = cvt4(f);
  }

  float mrow[8], lrow[8];
  v8f zero = {};
  v8f oacc[4];
  #pragma unroll
  for (int r = 0; r < 8; r++) { mrow[r] = -1e30f; lrow[r] = 0.f; }
  #pragma unroll
  for (int t = 0; t < 4; t++) oacc[t] = zero;

  for (int kt = 0; kt < n; kt += 32) {
    const bool kint = (kt + 32 <= n);
    v8f s0 = zero, s1 = zero;
    for (int hc = 0; hc < HH; hc += 128) {
      __syncthreads();
      if (kint) {
        #pragma unroll
        for (int i = 0; i < 4; i++) {
          int idx = tid + i * 256;               // 0..1023 vectors of 4
          int r = idx >> 5, c4 = (idx & 31) * 4;
          v4f f = *(const v4f*)&Km[(long)(kt + r) * HH + hc + c4];
          *(v4h*)&Ks[r][c4] = cvt4(f);
        }
      } else {
        #pragma unroll
        for (int i = 0; i < 16; i++) {
          int idx = tid + i * 256;
          int r = idx >> 7, c = idx & 127;
          float v = (kt + r < n) ? Km[(long)(kt + r) * HH + hc + c] : 0.f;
          Ks[r][c] = (half_t)v;
        }
      }
      __syncthreads();
      #pragma unroll
      for (int kk = 0; kk < 128; kk += 32) {
        v16h a  = frag_ld(&Qs[ln][hc + kk], hs);
        v16h b0 = frag_ld(&Ks[ln][kk], hs);
        v16h b1 = frag_ld(&Ks[16 + ln][kk], hs);
        s0 = WMMA_F16(a, b0, s0);
        s1 = WMMA_F16(a, b1, s1);
      }
    }

    // online softmax update for this 32-key tile
    #pragma unroll
    for (int r = 0; r < 8; r++) {
      int qi  = q0 + r + 8 * hs;
      int kj0 = kt + ln;
      int kj1 = kt + 16 + ln;
      float e0 = s0[r] * scale, e1 = s1[r] * scale;
      bool ok0 = (kj0 < n);
      if (ok0) ok0 = transposed ? (mask[(long)kj0 * n + qi] != 0)
                                : (mask[(long)qi * n + kj0] != 0);
      bool ok1 = (kj1 < n);
      if (ok1) ok1 = transposed ? (mask[(long)kj1 * n + qi] != 0)
                                : (mask[(long)qi * n + kj1] != 0);
      e0 = ok0 ? e0 : -1e30f;
      e1 = ok1 ? e1 : -1e30f;
      float rm = fmaxf(e0, e1);
      rm = fmaxf(rm, __shfl_xor(rm, 8, 32));
      rm = fmaxf(rm, __shfl_xor(rm, 4, 32));
      rm = fmaxf(rm, __shfl_xor(rm, 2, 32));
      rm = fmaxf(rm, __shfl_xor(rm, 1, 32));
      float mnew = fmaxf(mrow[r], rm);
      float resc = __expf(mrow[r] - mnew);
      float p0 = (e0 <= -1e29f) ? 0.f : __expf(e0 - mnew);
      float p1 = (e1 <= -1e29f) ? 0.f : __expf(e1 - mnew);
      float rs = p0 + p1;
      rs += __shfl_xor(rs, 8, 32);
      rs += __shfl_xor(rs, 4, 32);
      rs += __shfl_xor(rs, 2, 32);
      rs += __shfl_xor(rs, 1, 32);
      lrow[r] = lrow[r] * resc + rs;
      mrow[r] = mnew;
      #pragma unroll
      for (int t = 0; t < 4; t++) oacc[t][r] *= resc;
      Ps[wave][r + 8 * hs][ln]      = (half_t)p0;
      Ps[wave][r + 8 * hs][16 + ln] = (half_t)p1;
    }

    // P (16x32, A layout) @ V tile (32 x 64 cols owned by this wave)
    v16h pa = frag_ld(&Ps[wave][ln][0], hs);
    #pragma unroll
    for (int t = 0; t < 4; t++) {
      int col = wave * 64 + t * 16 + ln;
      v16h vb;
      if (kint) {
        #pragma unroll
        for (int i = 0; i < 4; i++) {
          int kA = hs * 8 + 2 * i;
          int kB = 16 + kA;
          vb[2 * i]     = (half_t)V[(long)(kt + kA) * HH + col];
          vb[2 * i + 1] = (half_t)V[(long)(kt + kA + 1) * HH + col];
          vb[8 + 2 * i]     = (half_t)V[(long)(kt + kB) * HH + col];
          vb[8 + 2 * i + 1] = (half_t)V[(long)(kt + kB + 1) * HH + col];
        }
      } else {
        #pragma unroll
        for (int i = 0; i < 4; i++) {
          int kA = hs * 8 + 2 * i;
          int kB = 16 + kA;
          float v00 = (kt + kA < n)     ? V[(long)(kt + kA) * HH + col]     : 0.f;
          float v01 = (kt + kA + 1 < n) ? V[(long)(kt + kA + 1) * HH + col] : 0.f;
          float v10 = (kt + kB < n)     ? V[(long)(kt + kB) * HH + col]     : 0.f;
          float v11 = (kt + kB + 1 < n) ? V[(long)(kt + kB + 1) * HH + col] : 0.f;
          vb[2 * i]     = (half_t)v00;
          vb[2 * i + 1] = (half_t)v01;
          vb[8 + 2 * i]     = (half_t)v10;
          vb[8 + 2 * i + 1] = (half_t)v11;
        }
      }
      oacc[t] = WMMA_F16(pa, vb, oacc[t]);
    }
  }

  #pragma unroll
  for (int r = 0; r < 8; r++) {
    int qi = q0 + r + 8 * hs;
    float denom = fmaxf(lrow[r], 1e-20f);
    #pragma unroll
    for (int t = 0; t < 4; t++) {
      int col = wave * 64 + t * 16 + ln;
      O[(long)qi * HH + col] = oacc[t][r] / denom;
    }
  }
}

// ---------------------------------------------------------------------------
// Host-side orchestration helpers
// ---------------------------------------------------------------------------
static inline int cdiv(long a, long b) { return (int)((a + b - 1) / b); }

static void gemm(hipStream_t s, const float* A, const float* B, const float* bias,
                 float* C, int M, int K, int Nc) {
  dim3 g(cdiv(Nc, 64), cdiv(M, 64));
  wmma_gemm_kernel<<<g, 256, 0, s>>>(A, B, bias, C, M, K, Nc);
}

static void gemv_elr(hipStream_t s, const float* h, const float* al, const float* ar,
                     float* el, float* er, int n, int d) {
  gemv_elr_kernel<<<n, 128, 0, s>>>(h, al, ar, el, er, d);
}

static void gat_edges(hipStream_t s, const int* src, const int* dst, int E,
                      const float* h, const float* el, const float* er,
                      float* acc, int n, int d,
                      unsigned* mbuf, float* sbuf, float* exbuf) {
  fill_enc_kernel<<<cdiv(n, 256), 256, 0, s>>>(mbuf, n, -1e30f);
  fill_f32_kernel<<<cdiv(n, 256), 256, 0, s>>>(sbuf, n, 0.f);
  long tot = (long)n * d;
  fill_f32_kernel<<<cdiv(tot, 256), 256, 0, s>>>(acc, tot, 0.f);
  int gb = cdiv(E, 256);
  edge_max_kernel<<<gb, 256, 0, s>>>(src, dst, el, er, mbuf, E);
  edge_expsum_kernel<<<gb, 256, 0, s>>>(src, dst, el, er, mbuf, sbuf, exbuf, E);
  edge_scatter_kernel<<<E, 256, 0, s>>>(src, dst, exbuf, sbuf, h, acc, d);
}

static void gat_epi(hipStream_t s, const float* A, const float* B, const float* bias,
                    float* out, long n, int d, int mode) {
  long tot = n * d;
  gat_epilogue_kernel<<<cdiv(tot, 256), 256, 0, s>>>(A, B, bias, out, tot, d, mode);
}

static void l2norm(hipStream_t s, const float* in, float* out, int n, int d) {
  l2norm_kernel<<<n, 128, 0, s>>>(in, out, d);
}

// ---------------------------------------------------------------------------
// kernel_launch: full forward pass
// ---------------------------------------------------------------------------
extern "C" void kernel_launch(void* const* d_in, const int* in_sizes, int n_in,
                              void* d_out, int out_size, void* d_ws, size_t ws_size,
                              hipStream_t stream) {
  (void)in_sizes; (void)n_in; (void)out_size; (void)ws_size;
  hipStream_t s = stream;

  // --- input mapping (setup_inputs() insertion order, depth-first) ---
  const float* W1   = (const float*)d_in[0];   // gat1_o1 [3000,512]
  const float* al1  = (const float*)d_in[1];
  const float* ar1  = (const float*)d_in[2];
  const float* b1   = (const float*)d_in[3];
  const float* W2   = (const float*)d_in[4];   // gat1_o2 [1000,512]
  const float* al2  = (const float*)d_in[5];
  const float* ar2  = (const float*)d_in[6];
  const float* b2   = (const float*)d_in[7];
  const float* Wg2  = (const float*)d_in[8];   // gat2 [512,128]
  const float* alg2 = (const float*)d_in[9];
  const float* arg2 = (const float*)d_in[10];
  const float* bg2  = (const float*)d_in[11];
  const float* W3a  = (const float*)d_in[12];  // gat3_o1 [128,512]
  const float* al3a = (const float*)d_in[13];
  const float* ar3a = (const float*)d_in[14];
  const float* b3a  = (const float*)d_in[15];
  const float* W3b  = (const float*)d_in[16];  // gat3_o2 [128,512]
  const float* al3b = (const float*)d_in[17];
  const float* ar3b = (const float*)d_in[18];
  const float* b3b  = (const float*)d_in[19];
  const float* W4a  = (const float*)d_in[20];  // gat4_o1 [512,3000]
  const float* al4a = (const float*)d_in[21];
  const float* ar4a = (const float*)d_in[22];
  const float* b4a  = (const float*)d_in[23];
  const float* W4b  = (const float*)d_in[24];  // gat4_o2 [512,1000]
  const float* al4b = (const float*)d_in[25];
  const float* ar4b = (const float*)d_in[26];
  const float* b4b  = (const float*)d_in[27];
  const float* Wq   = (const float*)d_in[28];
  const float* bq   = (const float*)d_in[29];
  const float* Wk   = (const float*)d_in[30];
  const float* bk   = (const float*)d_in[31];
  const float* Wv1  = (const float*)d_in[32];
  const float* bv1  = (const float*)d_in[33];
  const float* Wv2  = (const float*)d_in[34];
  const float* bv2  = (const float*)d_in[35];
  const float* Wo1  = (const float*)d_in[36];
  const float* bo1  = (const float*)d_in[37];
  const float* Wo2  = (const float*)d_in[38];
  const float* bo2  = (const float*)d_in[39];
  const float* feat1 = (const float*)d_in[40]; // [6000,3000]
  const float* feat2 = (const float*)d_in[41]; // [6000,1000]
  const unsigned char* mask = (const unsigned char*)d_in[42]; // [6000,6000] bool
  const int* src1  = (const int*)d_in[43];
  const int* dst1  = (const int*)d_in[44];
  const int* src2  = (const int*)d_in[45];
  const int* dst2  = (const int*)d_in[46];
  const int* fsrc1 = (const int*)d_in[47];
  const int* fdst1 = (const int*)d_in[48];
  const int* fsrc2 = (const int*)d_in[49];
  const int* fdst2 = (const int*)d_in[50];
  const int* csrc  = (const int*)d_in[51];
  const int* cdst  = (const int*)d_in[52];

  // --- workspace arena (floats) ---
  const long NH  = (long)NN * HH;        // 3,072,000
  const long NL2 = 2L * NN * LL;         // 1,536,000
  float* ws = (float*)d_ws;
  float* H1PRE = ws + 0 * NH;
  float* H2PRE = ws + 1 * NH;
  float* GA    = ws + 2 * NH;
  float* GB    = ws + 3 * NH;
  float* H1    = ws + 4 * NH;
  float* H2    = ws + 5 * NH;
  float* H1N   = ws + 6 * NH;
  float* H2N   = ws + 7 * NH;
  float* V1    = ws + 8 * NH;
  float* V2    = ws + 9 * NH;
  float* A1    = ws + 10 * NH;
  float* A2    = ws + 11 * NH;
  float* F12   = ws + 12 * NH;           // 2*NH (concat f1,f2)
  float* RPRE  = ws + 14 * NH;           // N*D1 = 18,000,000
  float* MU    = RPRE + (long)NN * DD1;
  float* HZ    = MU + NL2;
  float* EL    = HZ + NL2;
  float* ER    = EL + 2 * NN;
  unsigned* MB = (unsigned*)(ER + 2 * NN);
  float* SB    = (float*)(MB + 2 * NN);
  float* EX    = SB + 2 * NN;

  // slot aliases (sequential reuse)
  float* Qb   = H1PRE;
  float* Kb   = H2PRE;
  float* ATT1 = H1;
  float* ATT2 = H2;
  float* TPRE = A1;
  float* U    = A2;
  float* HC1  = F12;
  float* HC2  = F12 + NH;
  float* HF1  = V1;
  float* HF2  = V2;
  float* CPRE = A1;

  // --- output layout ---
  float* out = (float*)d_out;
  float* O_MU1 = out;
  float* O_MU2 = out + (long)NN * LL;
  float* O_REC1 = out + 2L * NN * LL;
  float* O_REC2 = O_REC1 + (long)NN * DD1;
  float* O_CR1  = O_REC2 + (long)NN * DD2;
  float* O_CR2  = O_CR1 + (long)NN * DD1;
  float* O_MC1  = O_CR2 + (long)NN * DD2;
  float* O_MC2  = O_MC1 + (long)NN * LL;

  // ====================== Encoder ======================
  gemm(s, feat1, W1, nullptr, H1PRE, NN, DD1, HH);
  gemv_elr(s, H1PRE, al1, ar1, EL, ER, NN, HH);
  gat_edges(s, src1, dst1, E1, H1PRE, EL, ER, GA, NN, HH, MB, SB, EX);
  gat_edges(s, fsrc1, fdst1, E1, H1PRE, EL, ER, GB, NN, HH, MB, SB, EX);
  gat_epi(s, GA, GB, b1, H1, NN, HH, 2);

  gemm(s, feat2, W2, nullptr, H2PRE, NN, DD2, HH);
  gemv_elr(s, H2PRE, al2, ar2, EL, ER, NN, HH);
  gat_edges(s, src2, dst2, E1, H2PRE, EL, ER, GA, NN, HH, MB, SB, EX);
  gat_edges(s, fsrc2, fdst2, E1, H2PRE, EL, ER, GB, NN, HH, MB, SB, EX);
  gat_epi(s, GA, GB, b2, H2, NN, HH, 2);

  l2norm(s, H1, H1N, NN, HH);
  l2norm(s, H2, H2N, NN, HH);

  // ====================== Cross attention ======================
  gemm(s, H1, Wv1, bv1, V1, NN, HH, HH);
  gemm(s, H2, Wv2, bv2, V2, NN, HH, HH);
  gemm(s, H1, Wq, bq, Qb, NN, HH, HH);   // overwrites H1PRE slot (free)
  gemm(s, H2, Wk, bk, Kb, NN, HH, HH);
  flash_attn_kernel<<<NN / 16, 256, 0, s>>>(Qb, Kb, V2, mask, ATT1, NN, 0);
  flash_attn_kernel<<<NN / 16, 256, 0, s>>>(Kb, Qb, V1, mask, ATT2, NN, 1);
  gemm(s, ATT1, Wo1, bo1, A1, NN, HH, HH);
  gemm(s, ATT2, Wo2, bo2, A2, NN, HH, HH);
  l2norm(s, A1, A1, NN, HH);  // a1n in place
  l2norm(s, A2, A2, NN, HH);  // a2n in place

  combine_elu_kernel<<<cdiv(NH, 256), 256, 0, s>>>(H1N, A1, F12, NH);
  combine_elu_kernel<<<cdiv(NH, 256), 256, 0, s>>>(A2, H2N, F12 + NH, NH);

  // ====================== mu = gat2(concat, csrc/cdst) ======================
  gemm(s, F12, Wg2, nullptr, HZ, 2 * NN, HH, LL);
  gemv_elr(s, HZ, alg2, arg2, EL, ER, 2 * NN, LL);
  gat_edges(s, csrc, cdst, EC, HZ, EL, ER, GA, 2 * NN, LL, MB, SB, EX);
  gat_epi(s, GA, nullptr, bg2, MU, 2 * NN, LL, 0);
  l2norm(s, MU, O_MU1, NN, LL);
  l2norm(s, MU + (long)NN * LL, O_MU2, NN, LL);

  // ====================== Decoders ======================
  struct DecP {
    const float *W3, *al3, *ar3, *b3, *W4, *al4, *ar4, *b4;
    const int *src, *dst;
    const float* z;
    float* outp;
    int D;
  };
  DecP decs[4] = {
    { W3a, al3a, ar3a, b3a, W4a, al4a, ar4a, b4a, src1, dst1, O_MU1, O_REC1, DD1 },
    { W3b, al3b, ar3b, b3b, W4b, al4b, ar4b, b4b, src2, dst2, O_MU2, O_REC2, DD2 },
    { W3a, al3a, ar3a, b3a, W4a, al4a, ar4a, b4a, src2, dst2, O_MU2, O_CR1,  DD1 },
    { W3b, al3b, ar3b, b3b, W4b, al4b, ar4b, b4b, src1, dst1, O_MU1, O_CR2,  DD2 },
  };
  for (int i = 0; i < 4; i++) {
    DecP& p = decs[i];
    gemm(s, p.z, p.W3, nullptr, TPRE, NN, LL, HH);
    gemv_elr(s, TPRE, p.al3, p.ar3, EL, ER, NN, HH);
    gat_edges(s, p.src, p.dst, E1, TPRE, EL, ER, GA, NN, HH, MB, SB, EX);
    gat_epi(s, GA, nullptr, p.b3, U, NN, HH, 1);           // elu(gat + b3)
    gemm(s, U, p.W4, nullptr, RPRE, NN, HH, p.D);
    gemv_elr(s, RPRE, p.al4, p.ar4, EL, ER, NN, p.D);
    gat_edges(s, p.src, p.dst, E1, RPRE, EL, ER, p.outp, NN, p.D, MB, SB, EX);
    gat_epi(s, p.outp, nullptr, p.b4, p.outp, NN, p.D, 0); // + b4 in place
  }

  // ====================== Cross encoder ======================
  gemm(s, O_CR1, W1, nullptr, CPRE, NN, DD1, HH);
  gemv_elr(s, CPRE, al1, ar1, EL, ER, NN, HH);
  gat_edges(s, src2, dst2, E1, CPRE, EL, ER, GA, NN, HH, MB, SB, EX);
  gat_edges(s, fsrc2, fdst2, E1, CPRE, EL, ER, GB, NN, HH, MB, SB, EX);
  gat_epi(s, GA, GB, b1, HC1, NN, HH, 2);

  gemm(s, O_CR2, W2, nullptr, CPRE, NN, DD2, HH);
  gemv_elr(s, CPRE, al2, ar2, EL, ER, NN, HH);
  gat_edges(s, src1, dst1, E1, CPRE, EL, ER, GA, NN, HH, MB, SB, EX);
  gat_edges(s, fsrc1, fdst1, E1, CPRE, EL, ER, GB, NN, HH, MB, SB, EX);
  gat_epi(s, GA, GB, b2, HC2, NN, HH, 2);
  l2norm(s, HC1, HC1, NN, HH);
  l2norm(s, HC2, HC2, NN, HH);

  combine_elu_kernel<<<cdiv(NH, 256), 256, 0, s>>>(H1N, HC2, HF1, NH);
  combine_elu_kernel<<<cdiv(NH, 256), 256, 0, s>>>(HC1, H2N, HF2, NH);

  // mc1 = gat2(hf1, src1/dst1); mc2 = gat2(hf2, src2/dst2)
  gemm(s, HF1, Wg2, nullptr, HZ, NN, HH, LL);
  gemv_elr(s, HZ, alg2, arg2, EL, ER, NN, LL);
  gat_edges(s, src1, dst1, E1, HZ, EL, ER, GA, NN, LL, MB, SB, EX);
  gat_epi(s, GA, nullptr, bg2, O_MC1, NN, LL, 0);

  gemm(s, HF2, Wg2, nullptr, HZ, NN, HH, LL);
  gemv_elr(s, HZ, alg2, arg2, EL, ER, NN, LL);
  gat_edges(s, src2, dst2, E1, HZ, EL, ER, GA, NN, LL, MB, SB, EX);
  gat_epi(s, GA, nullptr, bg2, O_MC2, NN, LL, 0);
}
